// Sequence_80779744903257
// MI455X (gfx1250) — matrix operand order repaired and checked by reading, not compiled, not run on hardware
//
#include <hip/hip_runtime.h>
#include <hip/hip_bf16.h>

// ---------------------------------------------------------------------------
// Persistent 2-layer LSTM (B=256, T=1024, H=128, FUT=16) for gfx1250 (MI455X)
//   * 16 workgroups x 256 threads (8 waves, wave32). Each WG owns 16 batch rows
//     and runs the whole 1039-step recurrence with no grid sync.
//   * GEMMs via v_wmma_f32_16x16x32_f16: wave w computes gate tiles
//     {w, 8+w, 16+w, 24+w} (= i/f/g/o tile w) -> LSTM cell math is pure VALU
//     on the accumulators, c-state stays in VGPRs.
//   * h1/h2 exchanged through double-buffered padded LDS tiles (3 barriers/step)
//   * f16 weights live in L2 (prepped once per launch), fragments loaded as
//     32B vectors each step.
// ---------------------------------------------------------------------------

typedef _Float16 v16h __attribute__((ext_vector_type(16)));
typedef _Float16 v8h  __attribute__((ext_vector_type(8)));
typedef float    v8f  __attribute__((ext_vector_type(8)));

#define BATCH   256
#define TSEQ    1024
#define HID     128
#define GATES   512          // 4*HID
#define FUT     16
#define TOUT    (TSEQ + FUT - 1)   // 1039
#define BTILE   16
#define HPAD    136          // 128 + 8 f16 pad -> row stride 68 dwords, no bank conflict

// ---- fast transcendentals (lower to v_exp/v_rcp TRANS ops) ----------------
__device__ __forceinline__ float fsigmoid(float x) {
    return __builtin_amdgcn_rcpf(1.0f + __builtin_amdgcn_exp2f(-1.442695040888963f * x));
}
__device__ __forceinline__ float ftanh(float x) {
    float e = __builtin_amdgcn_exp2f(-2.885390081777927f * x);
    return (1.0f - e) * __builtin_amdgcn_rcpf(1.0f + e);
}

// ---- A fragment: h tile [16 x 128] f16 in LDS, K-chunk kc (K=32) ----------
// A operand, contiguous k order used here: lanes 0-15 row M=lane get K = kc*32 + 0..15;
// lanes 16-31 row M=lane-16 get K = kc*32 + 16..31 (matches B fragment).
__device__ __forceinline__ v16h load_afrag(const _Float16* __restrict__ hb,
                                           int lo, int hi, int kc) {
    const _Float16* row = hb + lo * HPAD + kc * 32 + hi * 16;
    v16h a;
    *((v8h*)&a)       = *(const v8h*)(row);
    *(((v8h*)&a) + 1) = *(const v8h*)(row + 8);
    return a;
}

// ---- B fragment: W row-major [512][128] f16 in global (L2-resident) -------
// B operand, contiguous k order used here: lanes 0-15 col N=lane hold K = kc*32 + 0..15,
// lanes 16-31 col N=lane-16 hold K = kc*32 + 16..31  -> one contiguous 32B load.
__device__ __forceinline__ v16h load_bfrag(const _Float16* __restrict__ W,
                                           int n, int hi, int kc) {
    return *(const v16h*)(W + n * HID + kc * 32 + hi * 16);
}

// ---------------------------------------------------------------------------
// Prep: f32 -> f16 weight copies + fused biases, once per launch.
// ---------------------------------------------------------------------------
__global__ void lstm_prep(const float* __restrict__ Whh1, const float* __restrict__ Wih2,
                          const float* __restrict__ Whh2,
                          const float* __restrict__ bih1, const float* __restrict__ bhh1,
                          const float* __restrict__ bih2, const float* __restrict__ bhh2,
                          _Float16* __restrict__ w1, _Float16* __restrict__ w2,
                          _Float16* __restrict__ w3,
                          float* __restrict__ b1, float* __restrict__ b2) {
    int i = blockIdx.x * blockDim.x + threadIdx.x;
    if (i < GATES * HID) {
        w1[i] = (_Float16)Whh1[i];
        w2[i] = (_Float16)Wih2[i];
        w3[i] = (_Float16)Whh2[i];
    }
    if (i < GATES) {
        b1[i] = bih1[i] + bhh1[i];
        b2[i] = bih2[i] + bhh2[i];
    }
}

// ---------------------------------------------------------------------------
// Persistent recurrence kernel.
// ---------------------------------------------------------------------------
__global__ __launch_bounds__(256, 1)
void lstm_persist(const float* __restrict__ input,
                  const float* __restrict__ Wih1,        // [512] (512x1)
                  const _Float16* __restrict__ Whh1h,    // [512][128] f16
                  const _Float16* __restrict__ Wih2h,
                  const _Float16* __restrict__ Whh2h,
                  const float* __restrict__ bias1,       // [512] fused
                  const float* __restrict__ bias2,
                  const float* __restrict__ Wlin,        // [128]
                  const float* __restrict__ blin,        // [1]
                  float* __restrict__ out) {             // [256][1039]
    __shared__ _Float16 h1s[2][BTILE * HPAD];
    __shared__ _Float16 h2s[2][BTILE * HPAD];
    __shared__ float    xbuf[BTILE];
    __shared__ float    wlin_s[HID];

    const int tid   = threadIdx.x;
    const int w     = tid >> 5;        // wave 0..7
    const int lane  = tid & 31;
    const int lo    = lane & 15;
    const int hi    = lane >> 4;
    const int bbase = blockIdx.x * BTILE;

    // one-time staging
    if (tid < HID) wlin_s[tid] = Wlin[tid];
    for (int i = tid; i < BTILE * HPAD; i += 256) {
        h1s[0][i] = (_Float16)0.0f; h1s[1][i] = (_Float16)0.0f;
        h2s[0][i] = (_Float16)0.0f; h2s[1][i] = (_Float16)0.0f;
    }
    if (tid < BTILE) xbuf[tid] = input[(bbase + tid) * TSEQ + 0];

    // per-lane hoisted constants: gate column g = (q*8 + w)*16 + lo
    float b1v[4], b2v[4], wi1v[4];
#pragma unroll
    for (int q = 0; q < 4; ++q) {
        int g  = (q * 8 + w) * 16 + lo;
        b1v[q] = bias1[g];
        b2v[q] = bias2[g];
        wi1v[q] = Wih1[g];
    }
    const float bl = *blin;

    float c1r[8], c2r[8];
#pragma unroll
    for (int r = 0; r < 8; ++r) { c1r[r] = 0.0f; c2r[r] = 0.0f; }

    __syncthreads();

    int p = 0;
    for (int t = 0; t < TOUT; ++t) {
        // -------- per-row x values (m = r + 8*hi) --------
        float xv[8];
#pragma unroll
        for (int r = 0; r < 8; ++r) xv[r] = xbuf[r + 8 * hi];

        // ================= layer 1 =================
        v8f acc[4];
#pragma unroll
        for (int q = 0; q < 4; ++q)
#pragma unroll
            for (int r = 0; r < 8; ++r)
                acc[q][r] = b1v[q] + xv[r] * wi1v[q];   // bias + x (x) W_ih1 rank-1

        v16h a1[4];
#pragma unroll
        for (int kc = 0; kc < 4; ++kc) a1[kc] = load_afrag(h1s[p], lo, hi, kc);

#pragma unroll
        for (int q = 0; q < 4; ++q) {
            const int n = (q * 8 + w) * 16 + lo;
#pragma unroll
            for (int kc = 0; kc < 4; ++kc) {
                v16h bf = load_bfrag(Whh1h, n, hi, kc);
                acc[q] = __builtin_amdgcn_wmma_f32_16x16x32_f16(
                    false, a1[kc], false, bf, (short)0, acc[q], false, false);
            }
        }

        // cell update: (vgpr r, lane) holds same (m, j) across i/f/g/o accs
        {
            _Float16* h1w = h1s[p ^ 1] + (16 * w + lo);
#pragma unroll
            for (int r = 0; r < 8; ++r) {
                float iv = acc[0][r], fv = acc[1][r], gv = acc[2][r], ov = acc[3][r];
                float c  = fsigmoid(fv) * c1r[r] + fsigmoid(iv) * ftanh(gv);
                c1r[r]   = c;
                float h  = fsigmoid(ov) * ftanh(c);
                h1w[(r + 8 * hi) * HPAD] = (_Float16)h;
            }
        }
        __syncthreads();

        // ================= layer 2 =================
#pragma unroll
        for (int q = 0; q < 4; ++q)
#pragma unroll
            for (int r = 0; r < 8; ++r) acc[q][r] = b2v[q];

        v16h a2[4], a3[4];
#pragma unroll
        for (int kc = 0; kc < 4; ++kc) {
            a2[kc] = load_afrag(h1s[p ^ 1], lo, hi, kc);  // h1 (new)
            a3[kc] = load_afrag(h2s[p],     lo, hi, kc);  // h2 (old)
        }

#pragma unroll
        for (int q = 0; q < 4; ++q) {
            const int n = (q * 8 + w) * 16 + lo;
#pragma unroll
            for (int kc = 0; kc < 4; ++kc) {
                v16h bf = load_bfrag(Wih2h, n, hi, kc);
                acc[q] = __builtin_amdgcn_wmma_f32_16x16x32_f16(
                    false, a2[kc], false, bf, (short)0, acc[q], false, false);
            }
#pragma unroll
            for (int kc = 0; kc < 4; ++kc) {
                v16h bf = load_bfrag(Whh2h, n, hi, kc);
                acc[q] = __builtin_amdgcn_wmma_f32_16x16x32_f16(
                    false, a3[kc], false, bf, (short)0, acc[q], false, false);
            }
        }

        {
            _Float16* h2w = h2s[p ^ 1] + (16 * w + lo);
#pragma unroll
            for (int r = 0; r < 8; ++r) {
                float iv = acc[0][r], fv = acc[1][r], gv = acc[2][r], ov = acc[3][r];
                float c  = fsigmoid(fv) * c2r[r] + fsigmoid(iv) * ftanh(gv);
                c2r[r]   = c;
                float h  = fsigmoid(ov) * ftanh(c);
                h2w[(r + 8 * hi) * HPAD] = (_Float16)h;
            }
        }
        __syncthreads();

        // ============== linear head + next-x feedback ==============
        if (tid < BTILE) {
            const _Float16* hrow = h2s[p ^ 1] + tid * HPAD;
            float s = bl;
#pragma unroll
            for (int k0 = 0; k0 < HID; k0 += 8) {
                v8h hv = *(const v8h*)(hrow + k0);
#pragma unroll
                for (int j = 0; j < 8; ++j) s += (float)hv[j] * wlin_s[k0 + j];
            }
            out[(bbase + tid) * TOUT + t] = s;
            if (t + 1 < TOUT)
                xbuf[tid] = (t + 1 < TSEQ) ? input[(bbase + tid) * TSEQ + (t + 1)] : s;
        }
        __syncthreads();

        p ^= 1;
    }
}

// ---------------------------------------------------------------------------
extern "C" void kernel_launch(void* const* d_in, const int* in_sizes, int n_in,
                              void* d_out, int out_size, void* d_ws, size_t ws_size,
                              hipStream_t stream) {
    const float* input = (const float*)d_in[0];
    const float* Wih1  = (const float*)d_in[1];   // [512,1]
    const float* Whh1  = (const float*)d_in[2];   // [512,128]
    const float* bih1  = (const float*)d_in[3];
    const float* bhh1  = (const float*)d_in[4];
    const float* Wih2  = (const float*)d_in[5];
    const float* Whh2  = (const float*)d_in[6];
    const float* bih2  = (const float*)d_in[7];
    const float* bhh2  = (const float*)d_in[8];
    const float* Wlin  = (const float*)d_in[9];   // [1,128]
    const float* blin  = (const float*)d_in[10];
    (void)in_sizes; (void)n_in; (void)out_size; (void)ws_size;

    char* ws = (char*)d_ws;
    _Float16* w1 = (_Float16*)(ws);                 // 131072 B
    _Float16* w2 = (_Float16*)(ws + 131072);        // 131072 B
    _Float16* w3 = (_Float16*)(ws + 262144);        // 131072 B
    float*    b1 = (float*)(ws + 393216);           // 2048 B
    float*    b2 = (float*)(ws + 395264);           // 2048 B

    lstm_prep<<<256, 256, 0, stream>>>(Whh1, Wih2, Whh2, bih1, bhh1, bih2, bhh2,
                                       w1, w2, w3, b1, b2);
    lstm_persist<<<BATCH / BTILE, 256, 0, stream>>>(input, Wih1, w1, w2, w3,
                                                    b1, b2, Wlin, blin, (float*)d_out);
}
